// SCNN_64372969832793
// MI455X (gfx1250) — matrix operand order repaired and checked by reading, not compiled, not run on hardware
//
#include <hip/hip_runtime.h>

typedef __attribute__((ext_vector_type(16))) _Float16 v16h;
typedef __attribute__((ext_vector_type(8)))  _Float16 v8h;
typedef __attribute__((ext_vector_type(8)))  float    v8f;

// A fragment (16x32 f16, MxK): lane = 16*khalf + m ; slot h -> K = (h<8?h:h+8)+8*khalf.
// With K contiguous in memory at p (= base + 8*khalf), that is p[0..7] and p[16..23].
static __device__ __forceinline__ v16h ldA(const _Float16* p) {
    v8h lo = *(const v8h*)p;
    v8h hi = *(const v8h*)(p + 16);
    return __builtin_shufflevector(lo, hi, 0,1,2,3,4,5,6,7,8,9,10,11,12,13,14,15);
}
// B fragment (32x16 f16, KxN): lane = 16*khalf + n ; slot h -> K = (h<8?h:h+8)+8*khalf.
// With K contiguous at p (= base + 8*khalf): p[0..7] and p[16..23].
static __device__ __forceinline__ v16h ldB(const _Float16* p) {
    v8h lo = *(const v8h*)p;
    v8h hi = *(const v8h*)(p + 16);
    return __builtin_shufflevector(lo, hi, 0,1,2,3,4,5,6,7,8,9,10,11,12,13,14,15);
}

// ---------------------------------------------------------------------------
// Implicit-GEMM 3x3 conv + bias*scale+shift + ReLU.
// Input: f16 NHWC, zero halo of 1: ah[b][(H+2)][(W+2)][Cp]  (Cp % 32 == 0).
// Weights: f16 wh[tap][co][Cp]. Output: fp32 NCHW.
// One wave computes 32 pixels x 32 out-channels (2x2 WMMA register block).
// ---------------------------------------------------------------------------
__global__ __launch_bounds__(32) void conv3x3_wmma(
    const _Float16* __restrict__ ah, const _Float16* __restrict__ wh,
    const float* __restrict__ bias, const float* __restrict__ scale,
    const float* __restrict__ shift, float* __restrict__ out,
    int Cp, int Cout, int H, int W)
{
    const int lane  = threadIdx.x;
    const int mrow  = lane & 15;
    const int khalf = lane >> 4;
    const int tilesPerRow = W >> 5;
    const int y   = blockIdx.x / tilesPerRow;
    const int x0  = (blockIdx.x % tilesPerRow) << 5;
    const int co0 = blockIdx.y << 5;
    const int b   = blockIdx.z;
    const int Wp  = W + 2;
    const _Float16* base = ah + (size_t)b * (H + 2) * Wp * Cp;

    v8f acc[2][2] = {};

    for (int tap = 0; tap < 9; ++tap) {
        const int ky = tap / 3, kx = tap % 3;
        const _Float16* a0 = base + ((size_t)(y + ky) * Wp + (x0 + mrow + kx)) * Cp + 8 * khalf;
        const _Float16* a1 = a0 + (size_t)16 * Cp;
        const _Float16* w0 = wh + ((size_t)tap * Cout + co0 + mrow) * Cp + 8 * khalf;
        const _Float16* w1 = w0 + (size_t)16 * Cp;
        for (int ci0 = 0; ci0 < Cp; ci0 += 32) {
            v16h A[2]  = { ldA(a0 + ci0), ldA(a1 + ci0) };
            v16h Bt[2] = { ldB(w0 + ci0), ldB(w1 + ci0) };
#pragma unroll
            for (int t = 0; t < 2; ++t)
#pragma unroll
                for (int u = 0; u < 2; ++u)
                    acc[t][u] = __builtin_amdgcn_wmma_f32_16x16x32_f16(
                        false, A[t], false, Bt[u], (short)0, acc[t][u], false, false);
        }
    }

    const int n = lane & 15;
#pragma unroll
    for (int u = 0; u < 2; ++u) {
        const int co = co0 + u * 16 + n;
        const float bco = bias[co], sco = scale[co], tco = shift[co];
        float* ob = out + (((size_t)b * Cout + co) * H + y) * W + x0;
#pragma unroll
        for (int t = 0; t < 2; ++t)
#pragma unroll
            for (int r = 0; r < 8; ++r) {
                float v = (acc[t][u][r] + bco) * sco + tco;
                ob[t * 16 + r + 8 * khalf] = v > 0.0f ? v : 0.0f;
            }
    }
}

// ---------------------------------------------------------------------------
// One message-passing scan step, in place on f16 tensor mph[b][40][72][512]
// (spatial 32x64 with 4-wide zero halo on both dims):
//   cur[line] += relu(conv1d_9(cur[prev]))
// vertical: scan over H (rows), conv along W; horizontal: scan over W, conv
// along H. Branch-free: taps land inside the halo.
// ---------------------------------------------------------------------------
__global__ __launch_bounds__(32) void mp_step_wmma(
    _Float16* __restrict__ mph, const _Float16* __restrict__ wh,
    int vertical, int line, int prev)
{
    const int C = 512, Hp = 40, Wp = 72;
    const int lane  = threadIdx.x;
    const int mrow  = lane & 15;
    const int khalf = lane >> 4;
    const int x0  = blockIdx.x << 5;
    const int co0 = blockIdx.y << 5;
    const int b   = blockIdx.z;
    _Float16* base = mph + (size_t)b * Hp * Wp * C;

    v8f acc[2][2] = {};

    for (int tap = 0; tap < 9; ++tap) {
        const _Float16* ap[2];
#pragma unroll
        for (int t = 0; t < 2; ++t) {
            const int pos = x0 + t * 16 + mrow + tap;  // padded coord (unpadded + tap-4 + 4)
            ap[t] = vertical ? base + ((size_t)(prev + 4) * Wp + pos) * C + 8 * khalf
                             : base + ((size_t)pos * Wp + (prev + 4)) * C + 8 * khalf;
        }
        const _Float16* w0 = wh + ((size_t)tap * C + co0 + mrow) * C + 8 * khalf;
        const _Float16* w1 = w0 + (size_t)16 * C;
        for (int ci0 = 0; ci0 < C; ci0 += 32) {
            v16h A[2]  = { ldA(ap[0] + ci0), ldA(ap[1] + ci0) };
            v16h Bt[2] = { ldB(w0 + ci0), ldB(w1 + ci0) };
#pragma unroll
            for (int t = 0; t < 2; ++t)
#pragma unroll
                for (int u = 0; u < 2; ++u)
                    acc[t][u] = __builtin_amdgcn_wmma_f32_16x16x32_f16(
                        false, A[t], false, Bt[u], (short)0, acc[t][u], false, false);
        }
    }

    const int n = lane & 15;
#pragma unroll
    for (int u = 0; u < 2; ++u) {
        const int co = co0 + u * 16 + n;
#pragma unroll
        for (int t = 0; t < 2; ++t)
#pragma unroll
            for (int r = 0; r < 8; ++r) {
                const int pos = x0 + t * 16 + r + 8 * khalf;
                size_t idx = vertical
                    ? ((size_t)(line + 4) * Wp + (pos + 4)) * C + co
                    : ((size_t)(pos + 4) * Wp + (line + 4)) * C + co;
                float v = acc[t][u][r];
                v = v > 0.0f ? v : 0.0f;
                base[idx] = (_Float16)((float)base[idx] + v);
            }
    }
}

// ---------------------------------------------------------------------------
// Packing / layout kernels
// ---------------------------------------------------------------------------
// fp32 NCHW -> f16 NHWC with 1-pixel zero halo and channel padding to Cp.
__global__ void pack_act(const float* __restrict__ in, _Float16* __restrict__ oh,
                         int total, int C, int Cp, int H, int W)
{
    int i = blockIdx.x * blockDim.x + threadIdx.x;
    if (i >= total) return;            // total = 2*(H+2)*(W+2)*Cp
    int c = i % Cp; int t = i / Cp;
    int x = t % (W + 2); t /= (W + 2);
    int y = t % (H + 2); int b = t / (H + 2);
    float v = 0.0f;
    if (c < C && y >= 1 && y <= H && x >= 1 && x <= W)
        v = in[(((size_t)b * C + c) * H + (y - 1)) * W + (x - 1)];
    oh[i] = (_Float16)v;
}

// fp32 (co, ci, taps) -> f16 [tap][co][ci padded to Cp]. Works for 3x3 convs
// (taps=9 from OIHW 3x3) and for MP 1D kernels ((512,512,9) flat).
__global__ void pack_wgt(const float* __restrict__ w, _Float16* __restrict__ oh,
                         int total, int Cout, int Cin, int Cp)
{
    int i = blockIdx.x * blockDim.x + threadIdx.x;
    if (i >= total) return;            // total = 9*Cout*Cp
    int ci = i % Cp; int t = i / Cp;
    int co = t % Cout; int tap = t / Cout;
    float v = (ci < Cin) ? w[((size_t)co * Cin + ci) * 9 + tap] : 0.0f;
    oh[i] = (_Float16)v;
}

// fp32 NCHW (512,32,64) -> f16 [b][40][72][512] with 4-halo zeros.
__global__ void mp_init(const float* __restrict__ x7, _Float16* __restrict__ mph, int total)
{
    int i = blockIdx.x * blockDim.x + threadIdx.x;
    if (i >= total) return;            // total = 2*40*72*512
    int c = i % 512; int t = i / 512;
    int x = t % 72; t /= 72;
    int y = t % 40; int b = t / 40;
    float v = 0.0f;
    if (y >= 4 && y < 36 && x >= 4 && x < 68)
        v = x7[(((size_t)b * 512 + c) * 32 + (y - 4)) * 64 + (x - 4)];
    mph[i] = (_Float16)v;
}

// f16 padded NHWC -> fp32 NCHW (512,32,64).
__global__ void mp_fin(const _Float16* __restrict__ mph, float* __restrict__ x7, int total)
{
    int i = blockIdx.x * blockDim.x + threadIdx.x;
    if (i >= total) return;            // total = 2*512*32*64
    int x = i % 64; int t = i / 64;
    int y = t % 32; t /= 32;
    int c = t % 512; int b = t / 512;
    x7[i] = (float)mph[(((size_t)b * 40 + y + 4) * 72 + (x + 4)) * 512 + c];
}

// ---------------------------------------------------------------------------
// Small helper kernels (unchanged math)
// ---------------------------------------------------------------------------
__global__ void maxpool2(const float* __restrict__ in, float* __restrict__ out,
                         int total, int C, int Ho, int Wo)
{
    int i = blockIdx.x * blockDim.x + threadIdx.x;
    if (i >= total) return;
    int x = i % Wo; int t = i / Wo;
    int y = t % Ho; t /= Ho;
    int c = t % C;  int b = t / C;
    int W = Wo * 2, H = Ho * 2;
    const float* p = in + (((size_t)b * C + c) * H + 2 * y) * W + 2 * x;
    out[i] = fmaxf(fmaxf(p[0], p[1]), fmaxf(p[W], p[W + 1]));
}

__global__ void convt2x2(const float* __restrict__ in, const float* __restrict__ wt,
                         const float* __restrict__ bias, float* __restrict__ out,
                         int total, int Cin, int Cout, int Hin, int Win, int CT)
{
    int i = blockIdx.x * blockDim.x + threadIdx.x;
    if (i >= total) return;  // total = B*Cout*Hout*Wout
    int Ho = Hin * 2, Wo = Win * 2;
    int ox = i % Wo; int t = i / Wo;
    int oy = t % Ho; t /= Ho;
    int co = t % Cout; int b = t / Cout;
    int ix = ox >> 1, iy = oy >> 1, dx = ox & 1, dy = oy & 1;
    const float* ip = in + ((size_t)b * Cin * Hin + iy) * Win + ix;
    const float* wp = wt + (size_t)co * 4 + dy * 2 + dx;
    const size_t istr = (size_t)Hin * Win;
    const size_t wstr = (size_t)Cout * 4;
    float acc = bias[co];
    for (int ci = 0; ci < Cin; ++ci)
        acc += ip[(size_t)ci * istr] * wp[(size_t)ci * wstr];
    out[(((size_t)b * CT + co) * Ho + oy) * Wo + ox] = acc;
}

__global__ void copy_channels(const float* __restrict__ src, float* __restrict__ dst,
                              int total, int C, int HW, int CT, int coff)
{
    int i = blockIdx.x * blockDim.x + threadIdx.x;
    if (i >= total) return;   // total = B*C*HW
    int p = i % HW; int t = i / HW;
    int c = t % C;  int b = t / C;
    dst[((size_t)b * CT + coff + c) * HW + p] = src[((size_t)b * C + c) * HW + p];
}

__global__ void head1x1(const float* __restrict__ in, const float* __restrict__ w,
                        const float* __restrict__ b, float* __restrict__ out,
                        int total, int HW)
{
    int i = blockIdx.x * blockDim.x + threadIdx.x;
    if (i >= total) return;   // (2,5,HW)
    int p = i % HW; int t = i / HW;
    int co = t % 5; int bb = t / 5;
    const float* ip = in + (size_t)bb * 64 * HW + p;
    float acc = b[co];
    for (int ci = 0; ci < 64; ++ci) acc += ip[(size_t)ci * HW] * w[co * 64 + ci];
    out[i] = acc;
}

__global__ void softpool(const float* __restrict__ seg, float* __restrict__ p,
                         int total, int Ho, int Wo)
{
    int i = blockIdx.x * blockDim.x + threadIdx.x;
    if (i >= total) return;   // total = B*Ho*Wo
    int x = i % Wo; int t = i / Wo;
    int y = t % Ho; int b = t / Ho;
    int H = Ho * 2, W = Wo * 2;
    const size_t cstr = (size_t)H * W;
    float acc[5] = {0.f, 0.f, 0.f, 0.f, 0.f};
    for (int dy = 0; dy < 2; ++dy)
        for (int dx = 0; dx < 2; ++dx) {
            const float* sp = seg + ((size_t)b * 5 * H + (2 * y + dy)) * W + (2 * x + dx);
            float v[5], mx = -1e30f;
            for (int c = 0; c < 5; ++c) { v[c] = sp[c * cstr]; mx = fmaxf(mx, v[c]); }
            float s = 0.f;
            for (int c = 0; c < 5; ++c) { v[c] = __expf(v[c] - mx); s += v[c]; }
            float inv = 1.0f / s;
            for (int c = 0; c < 5; ++c) acc[c] += v[c] * inv;
        }
    for (int c = 0; c < 5; ++c)
        p[(((size_t)b * 5 + c) * Ho + y) * Wo + x] = acc[c] * 0.25f;
}

__global__ __launch_bounds__(256) void fc1_kernel(
    const float* __restrict__ flat, const float* __restrict__ W1,
    const float* __restrict__ b1, float* __restrict__ h, int N)
{
    __shared__ float red[256];
    const int j = blockIdx.x;
    const int b = blockIdx.y;
    const float* fp = flat + (size_t)b * N;
    float acc = 0.f;
    for (int i = threadIdx.x; i < N; i += 256)
        acc += fp[i] * W1[(size_t)i * 128 + j];
    red[threadIdx.x] = acc;
    __syncthreads();
    for (int s = 128; s > 0; s >>= 1) {
        if ((int)threadIdx.x < s) red[threadIdx.x] += red[threadIdx.x + s];
        __syncthreads();
    }
    if (threadIdx.x == 0) {
        float v = red[0] + b1[j];
        h[b * 128 + j] = v > 0.f ? v : 0.f;
    }
}

__global__ void fc2_kernel(const float* __restrict__ h, const float* __restrict__ W2,
                           const float* __restrict__ b2, float* __restrict__ out)
{
    int i = threadIdx.x;
    if (i >= 8) return;
    int b = i >> 2, j = i & 3;
    float acc = b2[j];
    for (int k = 0; k < 128; ++k) acc += h[b * 128 + k] * W2[k * 4 + j];
    out[i] = 1.0f / (1.0f + __expf(-acc));
}

// ---------------------------------------------------------------------------
// Host orchestration
// ---------------------------------------------------------------------------
extern "C" void kernel_launch(void* const* d_in, const int* in_sizes, int n_in,
                              void* d_out, int out_size, void* d_ws, size_t ws_size,
                              hipStream_t stream)
{
    (void)in_sizes; (void)n_in; (void)out_size; (void)ws_size;
    auto F = [&](int i) { return (const float*)d_in[i]; };
    const float* img = F(0);
    float* ws = (float*)d_ws;

    // fp32 buffers (element offsets; all 16B-aligned)
    float* x1   = ws;                       // (2,64,256,512)
    float* x1p  = x1   + 16777216;          // (2,64,128,256)
    float* x3   = x1p  +  4194304;          // (2,128,128,256)
    float* x3p  = x3   +  8388608;          // (2,128,64,128)
    float* x5   = x3p  +  2097152;          // (2,256,64,128)
    float* x5p  = x5   +  4194304;          // (2,256,32,64)
    float* x7   = x5p  +  1048576;          // (2,512,32,64)
    float* cat  = x7   +  2097152;          // max (2,128,256,512)
    float* tmp  = cat  + 33554432;          // max (2,64,256,512)
    float* dbuf = tmp  + 16777216;          // max (2,64,256,512)
    float* pbuf = dbuf + 16777216;          // (2,5,128,256)
    float* hbuf = pbuf +   327680;          // (2,128)
    // f16 staging buffers
    _Float16* stageA = (_Float16*)(hbuf + 256);          // max 2*258*514*128 halfs
    _Float16* stageW = (_Float16*)((float*)stageA + 16974336); // max 9*512*512 halfs
    _Float16* mpw    = (_Float16*)((float*)stageW + 1179648);  // 4 x 9*512*512 halfs
    _Float16* mph    = (_Float16*)((float*)mpw + 4718592);     // 2*40*72*512 halfs

    auto gb = [](int tot) { return (unsigned)((tot + 255) / 256); };

    auto dconv = [&](const float* in, int pb, float* t, float* o,
                     int Cin, int Cp, int Cout, int H, int W) {
        dim3 g((unsigned)(H * W / 32), (unsigned)(Cout / 32), 2u);
        int tp = 2 * (H + 2) * (W + 2) * Cp;
        pack_act<<<gb(tp), 256, 0, stream>>>(in, stageA, tp, Cin, Cp, H, W);
        int tw = 9 * Cout * Cp;
        pack_wgt<<<gb(tw), 256, 0, stream>>>(F(pb), stageW, tw, Cout, Cin, Cp);
        conv3x3_wmma<<<g, 32, 0, stream>>>(stageA, stageW, F(pb+1), F(pb+2), F(pb+3),
                                           t, Cp, Cout, H, W);
        tp = 2 * (H + 2) * (W + 2) * Cout;
        pack_act<<<gb(tp), 256, 0, stream>>>(t, stageA, tp, Cout, Cout, H, W);
        tw = 9 * Cout * Cout;
        pack_wgt<<<gb(tw), 256, 0, stream>>>(F(pb+4), stageW, tw, Cout, Cout, Cout);
        conv3x3_wmma<<<g, 32, 0, stream>>>(stageA, stageW, F(pb+5), F(pb+6), F(pb+7),
                                           o, Cout, Cout, H, W);
    };

    // ---------------- Encoder ----------------
    dconv(img, 1, tmp, x1, 3, 32, 64, 256, 512);
    { int tot = 2*64*128*256;  maxpool2<<<gb(tot),256,0,stream>>>(x1, x1p, tot, 64, 128, 256); }
    dconv(x1p, 9, tmp, x3, 64, 64, 128, 128, 256);
    { int tot = 2*128*64*128;  maxpool2<<<gb(tot),256,0,stream>>>(x3, x3p, tot, 128, 64, 128); }
    dconv(x3p, 17, tmp, x5, 128, 128, 256, 64, 128);
    { int tot = 2*256*32*64;   maxpool2<<<gb(tot),256,0,stream>>>(x5, x5p, tot, 256, 32, 64); }
    dconv(x5p, 25, tmp, x7, 256, 256, 512, 32, 64);

    // ---------------- Message passing ----------------
    { int tw = 9 * 512 * 512;
      for (int p = 0; p < 4; ++p)
          pack_wgt<<<gb(tw),256,0,stream>>>(F(33 + p), mpw + (size_t)p * tw, tw, 512, 512, 512);
      int ti = 2*40*72*512;
      mp_init<<<gb(ti),256,0,stream>>>(x7, mph, ti);

      const size_t WSTEP = (size_t)9 * 512 * 512;
      dim3 gv(64/32, 16, 2), gh(32/32, 16, 2);
      for (int y = 1; y < 32; ++y)                     // vertical forward
          mp_step_wmma<<<gv,32,0,stream>>>(mph, mpw + 0*WSTEP, 1, y, y-1);
      for (int y = 30; y >= 0; --y)                    // vertical reverse
          mp_step_wmma<<<gv,32,0,stream>>>(mph, mpw + 1*WSTEP, 1, y, y+1);
      for (int x = 1; x < 64; ++x)                     // horizontal forward
          mp_step_wmma<<<gh,32,0,stream>>>(mph, mpw + 2*WSTEP, 0, x, x-1);
      for (int x = 62; x >= 0; --x)                    // horizontal reverse
          mp_step_wmma<<<gh,32,0,stream>>>(mph, mpw + 3*WSTEP, 0, x, x+1);

      int tf = 2*512*32*64;
      mp_fin<<<gb(tf),256,0,stream>>>(mph, x7, tf);
    }

    // ---------------- Decoder ----------------
    { int tot = 2*256*64*128;   // up(512->256) @64x128, cat x5 -> 512ch, dd1 -> 256
      convt2x2<<<gb(tot),256,0,stream>>>(x7, F(37), F(38), cat, tot, 512, 256, 32, 64, 512);
      copy_channels<<<gb(tot),256,0,stream>>>(x5, cat, tot, 256, 64*128, 512, 256);
      dconv(cat, 43, tmp, dbuf, 512, 512, 256, 64, 128); }
    { int tot = 2*128*128*256;  // up(256->128) @128x256, cat x3 -> 256ch, dd2 -> 128
      convt2x2<<<gb(tot),256,0,stream>>>(dbuf, F(39), F(40), cat, tot, 256, 128, 64, 128, 256);
      copy_channels<<<gb(tot),256,0,stream>>>(x3, cat, tot, 128, 128*256, 256, 128);
      dconv(cat, 51, tmp, dbuf, 256, 256, 128, 128, 256); }
    { int tot = 2*64*256*512;   // up(128->64) @256x512, cat x1 -> 128ch, dd3 -> 64
      convt2x2<<<gb(tot),256,0,stream>>>(dbuf, F(41), F(42), cat, tot, 128, 64, 128, 256, 128);
      copy_channels<<<gb(tot),256,0,stream>>>(x1, cat, tot, 64, 256*512, 128, 64);
      dconv(cat, 59, tmp, dbuf, 128, 128, 64, 256, 512); }

    // ---------------- Head + softmax-pool + FC ----------------
    float* seg = (float*)d_out;                 // (2,5,256,512)
    { int tot = 2*5*256*512;
      head1x1<<<gb(tot),256,0,stream>>>(dbuf, F(67), F(68), seg, tot, 256*512); }
    { int tot = 2*128*256;
      softpool<<<gb(tot),256,0,stream>>>(seg, pbuf, tot, 128, 256); }
    fc1_kernel<<<dim3(128,2), 256, 0, stream>>>(pbuf, F(69), F(70), hbuf, 5*128*256);
    fc2_kernel<<<1, 32, 0, stream>>>(hbuf, F(71), F(72), seg + 2*5*256*512);
}
